// FFNCosineNorm_29291676959207
// MI455X (gfx1250) — compile-verified, matrix operand order checked by reading, not run on hardware
//
#include <hip/hip_runtime.h>

typedef __attribute__((ext_vector_type(16))) __bf16        v16bf;
typedef __attribute__((ext_vector_type(8)))  float         v8f;
typedef __attribute__((ext_vector_type(8)))  unsigned int  v8u;

#define BDIM   256
#define KDIM   256
#define NDIM   256
#define ROWS_PER_BLOCK 128          // 8 waves * 16 rows
#define LDS_PITCH 528               // 256 bf16 (512B) + 16B pad -> conflict-free b128 reads
#define LDS_BYTES (NDIM * LDS_PITCH)
#define EPSV 1e-10f

// Round-half-up f32->bf16 pair pack: two adds + one v_perm_b32.
__device__ __forceinline__ unsigned int pack2(float lo, float hi) {
    unsigned int ulo = __float_as_uint(lo) + 0x8000u;
    unsigned int uhi = __float_as_uint(hi) + 0x8000u;
    return __builtin_amdgcn_perm(uhi, ulo, 0x07060302u);
}

// ---------------------------------------------------------------------------
// Prep: W (fp32, row-major OUTxIN) -> bf16 workspace copy + 1/||w|| (fp32).
// One block per output row j; 256 threads cover K.
// ---------------------------------------------------------------------------
__global__ void cosnorm_prep(const float* __restrict__ W,
                             unsigned short* __restrict__ wbf,
                             float* __restrict__ rwlen) {
    const int j = blockIdx.x;
    const int t = threadIdx.x;
    const float v = W[j * KDIM + t];
    const unsigned int u = __float_as_uint(v) + 0x8000u;
    wbf[j * KDIM + t] = (unsigned short)(u >> 16);

    float ss = v * v;
    #pragma unroll
    for (int off = 16; off > 0; off >>= 1)
        ss += __shfl_xor(ss, off, 32);

    __shared__ float part[8];
    if ((t & 31) == 0) part[t >> 5] = ss;
    __syncthreads();
    if (t < 8) {
        float s = part[t];
        s += __shfl_xor(s, 4, 32);
        s += __shfl_xor(s, 2, 32);
        s += __shfl_xor(s, 1, 32);
        if (t == 0) rwlen[j] = __builtin_amdgcn_rsqf(s);   // 1/||w_j||
    }
}

// ---------------------------------------------------------------------------
// Main GEMM + cosine-normalize epilogue.
// Block: 256 threads (8 waves). Wave w handles rows [blk*128 + w*16, +16),
// all 256 columns (16 N-tiles), K swept in 8 steps of 32 (bf16 WMMA).
// ---------------------------------------------------------------------------
__global__ void __launch_bounds__(BDIM)
cosnorm_main(const float* __restrict__ x,
             const float* __restrict__ bias,
             const unsigned short* __restrict__ wbf,
             const float* __restrict__ rwlen,
             float* __restrict__ out) {
    extern __shared__ char smem[];          // only dyn-LDS in kernel -> base offset 0

    const int tid  = threadIdx.x;
    const int lane = tid & 31;
    const int wid  = tid >> 5;
    const int nlo  = lane & 15;
    const int hi   = lane >> 4;
    const int khalf = hi * 8;

    // ---- Stage bf16 W into LDS with padded pitch via async global->LDS DMA.
    // 256x256 bf16 = 8192 chunks of 16B; 32 chunks per thread.
    {
        #pragma unroll 4
        for (int i = 0; i < 32; ++i) {
            unsigned int c   = (unsigned int)tid + (unsigned int)i * 256u;
            unsigned int row = c >> 5;               // W row (output col)
            unsigned int col = c & 31u;              // 16B chunk within row
            unsigned int lds_off = row * LDS_PITCH + col * 16u;
            unsigned long long gaddr =
                (unsigned long long)(uintptr_t)(wbf + (size_t)c * 8u);
            asm volatile("global_load_async_to_lds_b128 %0, %1, off"
                         :: "v"(lds_off), "v"(gaddr) : "memory");
        }
        asm volatile("s_wait_asynccnt 0" ::: "memory");
    }
    __syncthreads();

    const int rowBase = blockIdx.x * ROWS_PER_BLOCK + wid * 16;
    const float* xrow = x + (size_t)(rowBase + nlo) * KDIM;

    v8f acc[16] = {};
    float sumsq = 0.0f;

    for (int ks = 0; ks < 8; ++ks) {
        const int k0 = ks * 32 + khalf;

        // ---- A fragment: lane holds row (nlo), K = k0..k0+7 and k0+16..k0+23
        const float4 f0 = *(const float4*)(xrow + k0);
        const float4 f1 = *(const float4*)(xrow + k0 + 4);
        const float4 f2 = *(const float4*)(xrow + k0 + 16);
        const float4 f3 = *(const float4*)(xrow + k0 + 20);

        sumsq = fmaf(f0.x, f0.x, sumsq); sumsq = fmaf(f0.y, f0.y, sumsq);
        sumsq = fmaf(f0.z, f0.z, sumsq); sumsq = fmaf(f0.w, f0.w, sumsq);
        sumsq = fmaf(f1.x, f1.x, sumsq); sumsq = fmaf(f1.y, f1.y, sumsq);
        sumsq = fmaf(f1.z, f1.z, sumsq); sumsq = fmaf(f1.w, f1.w, sumsq);
        sumsq = fmaf(f2.x, f2.x, sumsq); sumsq = fmaf(f2.y, f2.y, sumsq);
        sumsq = fmaf(f2.z, f2.z, sumsq); sumsq = fmaf(f2.w, f2.w, sumsq);
        sumsq = fmaf(f3.x, f3.x, sumsq); sumsq = fmaf(f3.y, f3.y, sumsq);
        sumsq = fmaf(f3.z, f3.z, sumsq); sumsq = fmaf(f3.w, f3.w, sumsq);

        v8u au;
        au[0] = pack2(f0.x, f0.y); au[1] = pack2(f0.z, f0.w);
        au[2] = pack2(f1.x, f1.y); au[3] = pack2(f1.z, f1.w);
        au[4] = pack2(f2.x, f2.y); au[5] = pack2(f2.z, f2.w);
        au[6] = pack2(f3.x, f3.y); au[7] = pack2(f3.z, f3.w);
        const v16bf afrag = __builtin_bit_cast(v16bf, au);

        // ---- 16 N-tiles: B fragment from LDS (two b128 reads), then WMMA.
        #pragma unroll
        for (int nt = 0; nt < 16; ++nt) {
            const int n = nt * 16 + nlo;
            const char* bp = smem + (size_t)n * LDS_PITCH + (size_t)k0 * 2;
            const uint4 b0 = *(const uint4*)(bp);
            const uint4 b1 = *(const uint4*)(bp + 32);
            v8u bu;
            bu[0] = b0.x; bu[1] = b0.y; bu[2] = b0.z; bu[3] = b0.w;
            bu[4] = b1.x; bu[5] = b1.y; bu[6] = b1.z; bu[7] = b1.w;
            const v16bf bfrag = __builtin_bit_cast(v16bf, bu);

            acc[nt] = __builtin_amdgcn_wmma_f32_16x16x32_bf16(
                false, afrag, false, bfrag, (short)0, acc[nt], false, false);
        }
    }

    // ---- 1/||x|| per row: combine the two K-halves, broadcast per row m.
    sumsq += __shfl_xor(sumsq, 16, 32);
    float rxl[8];
    #pragma unroll
    for (int v = 0; v < 8; ++v)
        rxl[v] = __builtin_amdgcn_rsqf(__shfl(sumsq, v + 8 * hi, 32));

    // ---- Epilogue: clamp(dot * (1/|x|) * (1/|w|), eps) + b ; store.
    #pragma unroll
    for (int nt = 0; nt < 16; ++nt) {
        const int n = nt * 16 + nlo;
        const float rwl = rwlen[n];
        const float bn  = bias[n];
        #pragma unroll
        for (int v = 0; v < 8; ++v) {
            float val = acc[nt][v] * (rxl[v] * rwl);
            val = fmaxf(val, EPSV) + bn;
            out[(size_t)(rowBase + v + 8 * hi) * NDIM + n] = val;
        }
    }
}

extern "C" void kernel_launch(void* const* d_in, const int* in_sizes, int n_in,
                              void* d_out, int out_size, void* d_ws, size_t ws_size,
                              hipStream_t stream) {
    const float* x    = (const float*)d_in[0];
    const float* W    = (const float*)d_in[1];
    const float* bias = (const float*)d_in[2];
    float* out = (float*)d_out;

    // workspace layout: [0,128KB) W as bf16, [128KB,+1KB) 1/w_len fp32
    unsigned short* wbf = (unsigned short*)d_ws;
    float* rwlen = (float*)((char*)d_ws + (size_t)NDIM * KDIM * sizeof(unsigned short));

    cosnorm_prep<<<NDIM, KDIM, 0, stream>>>(W, wbf, rwlen);

    const int rows = 65536;
    cosnorm_main<<<rows / ROWS_PER_BLOCK, BDIM, LDS_BYTES, stream>>>(
        x, bias, wbf, rwlen, out);
}
